// GraphProCCF_71966472012153
// MI455X (gfx1250) — matrix operand order repaired and checked by reading, not compiled, not run on hardware
//
#include <hip/hip_runtime.h>
#include <hip/hip_bf16.h>
#include <stdint.h>

typedef __attribute__((ext_vector_type(16))) _Float16 v16h;
typedef __attribute__((ext_vector_type(8)))  float    v8f;

#define NUM_USERS  100000
#define NUM_ITEMS  50000
#define N_NODES    150000
#define N_EDGES    2400000
#define EMB        64
#define BATCH      4096
#define GCN_LAYERS 3
#define INV_TAU    5.0f
#define SSL_LAMBDA 0.1f
#define REG_LAMBDA 1e-4f
// exp(dot/tau) == exp2(dot * LOG2E_OVER_TAU); folded into A-matrix values so
// the GEMM inner loop needs zero VALU muls before v_exp_f32.
#define LOG2E_OVER_TAU 7.2134752044448170f   // (1/0.2) * log2(e)

static constexpr size_t N64 = (size_t)N_NODES * EMB;        // 9,600,000 floats
static constexpr int    NBLK = BATCH / 16;                  // 256 row/col blocks
static constexpr int    FRAG_HALVES = NBLK * 2 * 32 * 16;   // per-matrix fragment halves

// ---------------------------------------------------------------- utilities
__device__ __forceinline__ float wsum32(float x) {
    x += __shfl_xor(x, 1, 32);
    x += __shfl_xor(x, 2, 32);
    x += __shfl_xor(x, 4, 32);
    x += __shfl_xor(x, 8, 32);
    x += __shfl_xor(x, 16, 32);
    return x;
}

// ------------------------------------------------- e = acc = emb, enext = 0
__global__ void gpc_init_kernel(const float* __restrict__ uemb,
                                const float* __restrict__ iemb,
                                float* __restrict__ e,
                                float* __restrict__ acc,
                                float* __restrict__ enext) {
    size_t i = (size_t)blockIdx.x * blockDim.x + threadIdx.x;   // float4 index
    size_t n4 = N64 / 4;
    if (i >= n4) return;
    size_t base = i * 4;
    size_t node = base >> 6;
    size_t d    = base & 63;
    const float* src = (node < (size_t)NUM_USERS)
                         ? (uemb + node * EMB + d)
                         : (iemb + (node - NUM_USERS) * EMB + d);
    float4 v = *(const float4*)src;
    ((float4*)e)[i]   = v;
    ((float4*)acc)[i] = v;
    ((float4*)enext)[i] = make_float4(0.f, 0.f, 0.f, 0.f);
}

__global__ void gpc_zero_scal_kernel(float* __restrict__ scal) {
    if (threadIdx.x < 8) scal[threadIdx.x] = 0.f;
}

// ----------------------------------- one wave per edge: out[row] += v*e[col]
__global__ void gpc_spmm_kernel(const int* __restrict__ row,
                                const int* __restrict__ col,
                                const float* __restrict__ val,
                                const float* __restrict__ e,
                                float* __restrict__ out, int nE) {
    int edge = blockIdx.x * (blockDim.x >> 5) + (threadIdx.x >> 5);
    if (edge >= nE) return;
    int lane = threadIdx.x & 31;
    int r = row[edge];
    int c = col[edge];
    float v = val[edge];
    float2 s = ((const float2*)(e + (size_t)c * EMB))[lane];
    float* dst = out + (size_t)r * EMB + lane * 2;
    atomicAdd(dst,     v * s.x);
    atomicAdd(dst + 1, v * s.y);
}

// ------------------------------------------- acc += enext; ezero = 0 (swap)
__global__ void gpc_accum_zero_kernel(float* __restrict__ acc,
                                      const float* __restrict__ enext,
                                      float* __restrict__ ezero) {
    size_t i = (size_t)blockIdx.x * blockDim.x + threadIdx.x;
    size_t n4 = N64 / 4;
    if (i >= n4) return;
    float4 a = ((const float4*)acc)[i];
    float4 b = ((const float4*)enext)[i];
    a.x += b.x; a.y += b.y; a.z += b.z; a.w += b.w;
    ((float4*)acc)[i] = a;
    ((float4*)ezero)[i] = make_float4(0.f, 0.f, 0.f, 0.f);
}

// --------- one wave per batch element: dots, BPR, reg, diag, normalized f16
// un16 is pre-scaled by LOG2E_OVER_TAU so WMMA emits exp2-domain logits.
__global__ void gpc_gather_kernel(const int* __restrict__ user,
                                  const int* __restrict__ posi,
                                  const int* __restrict__ negi,
                                  const float* __restrict__ acc,
                                  _Float16* __restrict__ un16,
                                  _Float16* __restrict__ pn16,
                                  float* __restrict__ scal) {
    int b = blockIdx.x * (blockDim.x >> 5) + (threadIdx.x >> 5);
    if (b >= BATCH) return;
    int lane = threadIdx.x & 31;
    int iu = user[b];
    int ip = posi[b] + NUM_USERS;
    int in_ = negi[b] + NUM_USERS;
    float2 u = ((const float2*)(acc + (size_t)iu * EMB))[lane];
    float2 p = ((const float2*)(acc + (size_t)ip * EMB))[lane];
    float2 n = ((const float2*)(acc + (size_t)in_ * EMB))[lane];
    const float q = 1.0f / (GCN_LAYERS + 1);   // mean over stacked layers
    u.x *= q; u.y *= q; p.x *= q; p.y *= q; n.x *= q; n.y *= q;

    float dup = wsum32(u.x * p.x + u.y * p.y);
    float dun = wsum32(u.x * n.x + u.y * n.y);
    float duu = wsum32(u.x * u.x + u.y * u.y);
    float dpp = wsum32(p.x * p.x + p.y * p.y);
    float dnn = wsum32(n.x * n.x + n.y * n.y);

    float inu = rsqrtf(duu);
    float inp = rsqrtf(dpp);
    float inu_s = inu * LOG2E_OVER_TAU;        // fold 1/tau * log2(e) into A

    _Float16* up = un16 + (size_t)b * EMB + lane * 2;
    up[0] = (_Float16)(u.x * inu_s);
    up[1] = (_Float16)(u.y * inu_s);
    _Float16* pp = pn16 + (size_t)b * EMB + lane * 2;
    pp[0] = (_Float16)(p.x * inp);
    pp[1] = (_Float16)(p.y * inp);

    if (lane == 0) {
        float y = dun - dup;                      // -(pos_s - neg_s)
        float bpr = fmaxf(y, 0.f) + log1pf(__expf(-fabsf(y)));  // softplus
        atomicAdd(scal + 0, bpr);
        atomicAdd(scal + 1, duu + dpp + dnn);
        atomicAdd(scal + 2, dup * inu * inp * INV_TAU);         // diag logit
    }
}

// ---------------- pack row-major f16 [4096][64] into WMMA fragment layouts
// A 16x32 f16 (ISA 7.12.2): lane=16h+m; VGPR p pair = K {2p+8h | p<4 ; 16+2(p-4)+8h}
__global__ void gpc_pack_a_kernel(const _Float16* __restrict__ src,
                                  _Float16* __restrict__ frag) {
    int t = blockIdx.x * blockDim.x + threadIdx.x;  // one half-pair per thread
    if (t >= NBLK * 2 * 32 * 8) return;
    int p    = t & 7;
    int lane = (t >> 3) & 31;
    int kc   = (t >> 8) & 1;
    int blk  = t >> 9;
    int h = lane >> 4;
    int m = lane & 15;
    int k = kc * 32 + ((p < 4) ? (8 * h + 2 * p) : (16 + 8 * h + 2 * (p - 4)));
    const _Float16* s = src + (size_t)(blk * 16 + m) * EMB + k;
    _Float16* d = frag + (size_t)t * 2;
    d[0] = s[0];
    d[1] = s[1];
}

// B 32x16 f16: lane=16h+n (n = column = pn row in block); VGPR p pair = K {2p+8h | p<4 ; 16+2(p-4)+8h}
__global__ void gpc_pack_b_kernel(const _Float16* __restrict__ src,
                                  _Float16* __restrict__ frag) {
    int t = blockIdx.x * blockDim.x + threadIdx.x;
    if (t >= NBLK * 2 * 32 * 8) return;
    int p    = t & 7;
    int lane = (t >> 3) & 31;
    int kc   = (t >> 8) & 1;
    int blk  = t >> 9;
    int h = lane >> 4;
    int n = lane & 15;
    int k = kc * 32 + ((p < 4) ? (8 * h + 2 * p) : (16 + 8 * h + 2 * (p - 4)));
    const _Float16* s = src + (size_t)(blk * 16 + n) * EMB + k;
    _Float16* d = frag + (size_t)t * 2;
    d[0] = s[0];
    d[1] = s[1];
}

// -------- one wave per 16-row strip: WMMA logits + streaming sum(exp) rows
__global__ void gpc_infonce_kernel(const _Float16* __restrict__ afrag,
                                   const _Float16* __restrict__ bfrag,
                                   float* __restrict__ scal) {
    int w = blockIdx.x * (blockDim.x >> 5) + (threadIdx.x >> 5);
    if (w >= NBLK) return;
    int lane = threadIdx.x & 31;

    const v16h a0 = *(const v16h*)(afrag + ((size_t)(w * 2 + 0) * 32 + lane) * 16);
    const v16h a1 = *(const v16h*)(afrag + ((size_t)(w * 2 + 1) * 32 + lane) * 16);

    float sums[8] = {0.f, 0.f, 0.f, 0.f, 0.f, 0.f, 0.f, 0.f};

    for (int j = 0; j < NBLK; ++j) {
        v16h b0 = *(const v16h*)(bfrag + ((size_t)(j * 2 + 0) * 32 + lane) * 16);
        v16h b1 = *(const v16h*)(bfrag + ((size_t)(j * 2 + 1) * 32 + lane) * 16);
        v8f c = {};
        c = __builtin_amdgcn_wmma_f32_16x16x32_f16(false, a0, false, b0,
                                                   (short)0, c, false, false);
        c = __builtin_amdgcn_wmma_f32_16x16x32_f16(false, a1, false, b1,
                                                   (short)0, c, false, false);
        // logits arrive already in exp2 domain (scale folded into A);
        // |logit/tau| <= 5 so no overflow and no online-max pass is needed.
#pragma unroll
        for (int r = 0; r < 8; ++r) sums[r] += __builtin_amdgcn_exp2f(c[r]);
    }
    // reduce over the 16 lanes of each half-wave (columns of this lane group)
#pragma unroll
    for (int r = 0; r < 8; ++r) {
        float x = sums[r];
        x += __shfl_xor(x, 1, 32);
        x += __shfl_xor(x, 2, 32);
        x += __shfl_xor(x, 4, 32);
        x += __shfl_xor(x, 8, 32);
        sums[r] = x;
    }
    if ((lane & 15) == 0) {     // lane 0 -> rows 16w..16w+7, lane 16 -> +8..+15
        float acc = 0.f;
#pragma unroll
        for (int r = 0; r < 8; ++r) acc += __logf(sums[r]);
        atomicAdd(scal + 3, acc);
    }
}

__global__ void gpc_finalize_kernel(const float* __restrict__ scal,
                                    float* __restrict__ out) {
    if (threadIdx.x == 0 && blockIdx.x == 0) {
        const float invB = 1.0f / (float)BATCH;
        float bpr = scal[0] * invB;
        float reg = 0.5f * scal[1] * invB * REG_LAMBDA;
        float na  = (scal[3] - scal[2]) * invB * SSL_LAMBDA;
        out[0] = bpr + reg + na;
    }
}

// --------------------------------------------------------------------------
extern "C" void kernel_launch(void* const* d_in, const int* in_sizes, int n_in,
                              void* d_out, int out_size, void* d_ws, size_t ws_size,
                              hipStream_t stream) {
    (void)in_sizes; (void)n_in; (void)out_size; (void)ws_size;
    const int*   user = (const int*)d_in[0];
    const int*   posi = (const int*)d_in[1];
    const int*   negi = (const int*)d_in[2];
    const int*   arow = (const int*)d_in[3];
    const int*   acol = (const int*)d_in[4];
    const float* aval = (const float*)d_in[5];
    const float* uemb = (const float*)d_in[6];
    const float* iemb = (const float*)d_in[7];

    char* ws = (char*)d_ws;
    size_t off = 0;
    auto alloc = [&](size_t bytes) -> void* {
        void* p = ws + off;
        off = (off + bytes + 63) & ~(size_t)63;
        return p;
    };
    float*    e     = (float*)alloc(N64 * sizeof(float));
    float*    enext = (float*)alloc(N64 * sizeof(float));
    float*    acc   = (float*)alloc(N64 * sizeof(float));
    _Float16* un16  = (_Float16*)alloc((size_t)BATCH * EMB * sizeof(_Float16));
    _Float16* pn16  = (_Float16*)alloc((size_t)BATCH * EMB * sizeof(_Float16));
    _Float16* afrag = (_Float16*)alloc((size_t)FRAG_HALVES * sizeof(_Float16));
    _Float16* bfrag = (_Float16*)alloc((size_t)FRAG_HALVES * sizeof(_Float16));
    float*    scal  = (float*)alloc(8 * sizeof(float));

    // init node buffers + scalar accumulators
    {
        size_t n4 = N64 / 4;
        int tb = 256;
        int nb = (int)((n4 + tb - 1) / tb);
        gpc_init_kernel<<<nb, tb, 0, stream>>>(uemb, iemb, e, acc, enext);
    }
    gpc_zero_scal_kernel<<<1, 32, 0, stream>>>(scal);

    // 3 GCN layers: scatter-add SpMM (wave per edge), then acc += layer, swap
    float* src = e;
    float* dst = enext;
    for (int l = 0; l < GCN_LAYERS; ++l) {
        int epb = 256 / 32;  // 8 edges per 256-thread block
        int nb = (N_EDGES + epb - 1) / epb;
        gpc_spmm_kernel<<<nb, 256, 0, stream>>>(arow, acol, aval, src, dst, N_EDGES);
        size_t n4 = N64 / 4;
        int nb2 = (int)((n4 + 255) / 256);
        gpc_accum_zero_kernel<<<nb2, 256, 0, stream>>>(acc, dst, src);
        float* t = src; src = dst; dst = t;
    }

    // batch gather + losses + normalized f16 embeddings
    gpc_gather_kernel<<<BATCH / 8, 256, 0, stream>>>(user, posi, negi, acc,
                                                     un16, pn16, scal);

    // pack WMMA fragments (A for un, B for pn)
    {
        int total = NBLK * 2 * 32 * 8;
        int nb = (total + 255) / 256;
        gpc_pack_a_kernel<<<nb, 256, 0, stream>>>(un16, afrag);
        gpc_pack_b_kernel<<<nb, 256, 0, stream>>>(pn16, bfrag);
    }

    // 4096x4096x64 logits GEMM + streaming logsumexp (256 waves)
    gpc_infonce_kernel<<<NBLK / 4, 128, 0, stream>>>(afrag, bfrag, scal);

    gpc_finalize_kernel<<<1, 64, 0, stream>>>(scal, (float*)d_out);
}
